// SimpleMCADModel_18803366822142
// MI455X (gfx1250) — matrix operand order repaired and checked by reading, not compiled, not run on hardware
//
#include <hip/hip_runtime.h>
#include <hip/hip_bf16.h>
#include <math.h>

#define TT 16
#define NN 20000
#define EE 320000
#define BB 16
#define HH 64
#define ETOT (EE + NN)   // edges + self loops

typedef __attribute__((ext_vector_type(16))) _Float16 v16h;
typedef __attribute__((ext_vector_type(8)))  float    v8f;

__device__ __forceinline__ float lrelu_(float v){ return v > 0.f ? v : 0.2f * v; }
__device__ __forceinline__ float elu_(float v){ return v > 0.f ? v : (expf(v) - 1.f); }
__device__ __forceinline__ float sigm_(float v){ return 1.f / (1.f + expf(-v)); }

// float atomic max via signed-max / unsigned-min bit trick (works for mixed signs)
__device__ __forceinline__ void atomicMaxF(float* addr, float v){
  if (v >= 0.f) atomicMax((int*)addr, __float_as_int(v));
  else          atomicMin((unsigned int*)addr, __float_as_uint(v));
}

__global__ void fill_kernel(float* p, int n, float v){
  int t = blockIdx.x * blockDim.x + threadIdx.x;
  if (t < n) p[t] = v;
}

__device__ __forceinline__ void cvt4_(v16h& a, int base, float4 f){
  a[base + 0] = (_Float16)f.x;
  a[base + 1] = (_Float16)f.y;
  a[base + 2] = (_Float16)f.z;
  a[base + 3] = (_Float16)f.w;
}

// H[n_rows,64] = X[n_rows,FIN] @ W[FIN,64] (FIN zero-padded to mult of 32)
// f16 inputs, f32 accumulation via v_wmma_f32_16x16x32_f16.
// 128 threads = 4 waves; each wave owns a 16x64 output tile.
// W is staged in LDS pre-packed in per-lane fragment order so each B fragment
// is one aligned 32-byte LDS read (2x ds_load_b128).
template<int FIN>
__global__ void gemm_wmma_kernel(const float* __restrict__ X,
                                 const float* __restrict__ W,
                                 float* __restrict__ H,
                                 int n_rows){
  constexpr int KPAD = (FIN + 31) & ~31;      // 32 or 64
  constexpr int KS   = KPAD / 32;             // k-steps
  __shared__ v16h Wpack[KS * 2 * 64];         // [kstep][g][n] -> 16 halfs

  _Float16* wp = (_Float16*)Wpack;
  for (int idx = threadIdx.x; idx < KPAD * 64; idx += blockDim.x){
    int i  = idx & 15;
    int n  = (idx >> 4) & 63;
    int gg = (idx >> 10) & 1;
    int ks = idx >> 11;
    int k  = ks * 32 + ((i < 8) ? (gg * 8 + i) : (16 + gg * 8 + (i - 8)));
    wp[idx] = (k < FIN) ? (_Float16)W[k * 64 + n] : (_Float16)0.0f;
  }
  __syncthreads();

  int wave = threadIdx.x >> 5;
  int lane = threadIdx.x & 31;
  int g    = lane >> 4;        // lane group 0/1
  int ln   = lane & 15;
  int m0   = (blockIdx.x * 4 + wave) * 16;
  if (m0 >= n_rows) return;    // wave-uniform -> EXEC stays all-ones for WMMA

  const float* xrow = X + (size_t)(m0 + ln) * FIN;
  __builtin_prefetch(X + (size_t)(m0 + 64) * FIN, 0, 0); // next node tile

  v8f acc[4];
  #pragma unroll
  for (int t = 0; t < 4; ++t)
    #pragma unroll
    for (int j = 0; j < 8; ++j) acc[t][j] = 0.f;

  #pragma unroll
  for (int ks = 0; ks < KS; ++ks){
    const int kk = ks * 32;
    // A fragment: 16-bit A 16x32 layout
    //   a[0..7]  <- k = kk + g*8 + i
    //   a[8..15] <- k = kk + 16 + g*8 + (i-8)
    v16h a;
    if constexpr ((FIN & 31) == 0){
      // contiguous 8-float runs -> vector b128 loads, packed cvt
      const float* b0 = xrow + kk + g * 8;
      const float* b1 = xrow + kk + 16 + g * 8;
      float4 f0 = *(const float4*)(b0);
      float4 f1 = *(const float4*)(b0 + 4);
      float4 f2 = *(const float4*)(b1);
      float4 f3 = *(const float4*)(b1 + 4);
      cvt4_(a, 0, f0); cvt4_(a, 4, f1); cvt4_(a, 8, f2); cvt4_(a, 12, f3);
    } else {
      // branchless: clamped in-bounds load + select
      #pragma unroll
      for (int i = 0; i < 16; ++i){
        int k  = (i < 8) ? (kk + g * 8 + i) : (kk + 16 + g * 8 + (i - 8));
        int kc = k < FIN ? k : FIN - 1;
        float v = xrow[kc];
        a[i] = (k < FIN) ? (_Float16)v : (_Float16)0.0f;
      }
    }
    #pragma unroll
    for (int t = 0; t < 4; ++t){
      // B fragment: single 32-byte aligned LDS read in fragment order
      v16h b = Wpack[(ks * 2 + g) * 64 + t * 16 + ln];
      acc[t] = __builtin_amdgcn_wmma_f32_16x16x32_f16(
          false, a, false, b, (short)0, acc[t], false, false);
    }
  }

  // D layout: VGPR r -> row = m0 + r + 8*g, col = t*16 + ln
  #pragma unroll
  for (int t = 0; t < 4; ++t)
    #pragma unroll
    for (int r = 0; r < 8; ++r)
      H[(size_t)(m0 + r + 8 * g) * 64 + t * 16 + ln] = acc[t][r];
}

// per (node, head): a_s = <h[n,head,:], att_s[head,:]>, same for a_d
__global__ void att_kernel(const float* __restrict__ h,
                           const float* __restrict__ att_s,
                           const float* __restrict__ att_d,
                           float* a_s, float* a_d, int n, int heads, int C){
  int tid = blockIdx.x * blockDim.x + threadIdx.x;
  if (tid >= n * heads) return;
  int node = tid / heads, hh = tid % heads;
  const float* row = h + (size_t)node * heads * C + hh * C;
  float ss = 0.f, dd = 0.f;
  for (int c = 0; c < C; c += 4){
    float4 v = *(const float4*)(row + c);
    float4 s4 = *(const float4*)(att_s + hh * C + c);
    float4 d4 = *(const float4*)(att_d + hh * C + c);
    ss += v.x * s4.x + v.y * s4.y + v.z * s4.z + v.w * s4.w;
    dd += v.x * d4.x + v.y * d4.y + v.z * d4.z + v.w * d4.w;
  }
  a_s[tid] = ss;
  a_d[tid] = dd;
}

__global__ void edge_max_kernel(const int* __restrict__ src, const int* __restrict__ dst,
                                int Ereal, int Etot,
                                const float* __restrict__ a_s, const float* __restrict__ a_d,
                                float* emax, int heads){
  int tid = blockIdx.x * blockDim.x + threadIdx.x;
  if (tid >= Etot * heads) return;
  int e = tid / heads, h = tid % heads;
  int s, d;
  if (e < Ereal){ s = src[e]; d = dst[e]; } else { s = d = e - Ereal; }
  float v = lrelu_(a_s[s * heads + h] + a_d[d * heads + h]);
  atomicMaxF(&emax[d * heads + h], v);
}

__global__ void edge_exp_kernel(const int* __restrict__ src, const int* __restrict__ dst,
                                int Ereal, int Etot,
                                const float* __restrict__ a_s, const float* __restrict__ a_d,
                                const float* __restrict__ emax,
                                float* den, float* ebuf, int heads){
  int tid = blockIdx.x * blockDim.x + threadIdx.x;
  if (tid >= Etot * heads) return;
  int e = tid / heads, h = tid % heads;
  int s, d;
  if (e < Ereal){ s = src[e]; d = dst[e]; } else { s = d = e - Ereal; }
  float v = lrelu_(a_s[s * heads + h] + a_d[d * heads + h]);
  float ev = expf(v - emax[d * heads + h]);
  ebuf[tid] = ev;
  atomicAdd(&den[d * heads + h], ev);
}

// per (edge, 4 channels): out[dst, c..c+3] += h[src, c..c+3] * alpha(edge, head(c))
// 4-channel group never crosses a head boundary (C = 64/heads >= 32)
__global__ void edge_msg_kernel(const int* __restrict__ src, const int* __restrict__ dst,
                                int Ereal, int Etot,
                                const float* __restrict__ hfeat,
                                const float* __restrict__ ebuf,
                                const float* __restrict__ den,
                                float* out, int heads){
  int tid = blockIdx.x * blockDim.x + threadIdx.x;
  if (tid >= Etot * 16) return;
  int e  = tid >> 4;
  int c4 = (tid & 15) * 4;
  int s, d;
  if (e < Ereal){ s = src[e]; d = dst[e]; } else { s = d = e - Ereal; }
  int C  = 64 / heads;
  int hh = c4 / C;
  float alpha = ebuf[e * heads + hh] / (den[d * heads + hh] + 1e-16f);
  float4 hv = *(const float4*)(hfeat + (size_t)s * 64 + c4);
  float* o  = out + (size_t)d * 64 + c4;
  atomicAdd(o + 0, hv.x * alpha);
  atomicAdd(o + 1, hv.y * alpha);
  atomicAdd(o + 2, hv.z * alpha);
  atomicAdd(o + 3, hv.w * alpha);
}

__global__ void bias_act_kernel(float* p, const float* __restrict__ bias, int n4){
  int t = blockIdx.x * blockDim.x + threadIdx.x;
  if (t >= n4) return;
  int c4 = (t & 15) * 4;
  float4 v  = ((const float4*)p)[t];
  float4 bb = *(const float4*)(bias + c4);
  v.x = elu_(v.x + bb.x);
  v.y = elu_(v.y + bb.y);
  v.z = elu_(v.z + bb.z);
  v.w = elu_(v.w + bb.w);
  ((float4*)p)[t] = v;
}

__global__ void pool_cnt_kernel(const int* __restrict__ bat, float* cnt, int n){
  int t = blockIdx.x * blockDim.x + threadIdx.x;
  if (t < n) atomicAdd(&cnt[bat[t]], 1.f);
}

__global__ void pool_sum_kernel(const int* __restrict__ bat,
                                const float* __restrict__ h, float* emb, int n){
  int t = blockIdx.x * blockDim.x + threadIdx.x;
  if (t >= n * 16) return;
  int node = t >> 4;
  int c4   = (t & 15) * 4;
  int b    = bat[node];
  float4 v = *(const float4*)(h + (size_t)node * 64 + c4);
  float* o = emb + b * 64 + c4;
  atomicAdd(o + 0, v.x);
  atomicAdd(o + 1, v.y);
  atomicAdd(o + 2, v.z);
  atomicAdd(o + 3, v.w);
}

// 16-step GRU over [B=16, H=64]; single block, state in LDS
__global__ void gru_kernel(const float* __restrict__ emb, const float* __restrict__ cnt,
                           const float* __restrict__ Wih, const float* __restrict__ Whh,
                           const float* __restrict__ bih, const float* __restrict__ bhh,
                           float* hout){
  __shared__ float xs[BB * HH];
  __shared__ float hs[BB * HH];
  __shared__ float gx[BB * 3 * HH];
  __shared__ float gh[BB * 3 * HH];
  int tid = threadIdx.x; // 256
  for (int i = tid; i < BB * HH; i += 256) hs[i] = 0.f;
  __syncthreads();
  for (int t = 0; t < TT; ++t){
    for (int i = tid; i < BB * HH; i += 256){
      int b = i >> 6;
      float c = cnt[t * BB + b];
      xs[i] = emb[((size_t)t * BB + b) * HH + (i & 63)] / fmaxf(c, 1.f);
    }
    __syncthreads();
    for (int i = tid; i < BB * 3 * HH; i += 256){
      int b = i / (3 * HH), j = i % (3 * HH);
      float ax = bih[j], ah = bhh[j];
      const float* wi = Wih + (size_t)j * HH;
      const float* wh = Whh + (size_t)j * HH;
      for (int c = 0; c < HH; c += 4){
        float4 w4 = *(const float4*)(wi + c);
        float4 v4 = *(const float4*)(wh + c);
        ax += xs[b*HH+c]*w4.x + xs[b*HH+c+1]*w4.y + xs[b*HH+c+2]*w4.z + xs[b*HH+c+3]*w4.w;
        ah += hs[b*HH+c]*v4.x + hs[b*HH+c+1]*v4.y + hs[b*HH+c+2]*v4.z + hs[b*HH+c+3]*v4.w;
      }
      gx[i] = ax;
      gh[i] = ah;
    }
    __syncthreads();
    for (int i = tid; i < BB * HH; i += 256){
      int b = i >> 6, c = i & 63;
      float xr = gx[b * 192 + c], xz = gx[b * 192 + 64 + c], xn = gx[b * 192 + 128 + c];
      float hr = gh[b * 192 + c], hz = gh[b * 192 + 64 + c], hn = gh[b * 192 + 128 + c];
      float r  = sigm_(xr + hr);
      float z  = sigm_(xz + hz);
      float nn = tanhf(xn + r * hn);
      hs[i] = (1.f - z) * nn + z * hs[i];
    }
    __syncthreads();
  }
  for (int i = tid; i < BB * HH; i += 256) hout[i] = hs[i];
}

// relu(hn @ Wc1 + bc1) @ Wc2 + bc2 -> [B,1]
__global__ void head_kernel(const float* __restrict__ hn,
                            const float* __restrict__ Wc1, const float* __restrict__ bc1,
                            const float* __restrict__ Wc2, const float* __restrict__ bc2,
                            float* out){
  __shared__ float hid[BB * 32];
  int tid = threadIdx.x; // 512
  if (tid < BB * 32){
    int b = tid >> 5, j = tid & 31;
    float acc = bc1[j];
    for (int c = 0; c < HH; ++c) acc += hn[b * HH + c] * Wc1[c * 32 + j];
    hid[tid] = fmaxf(acc, 0.f);
  }
  __syncthreads();
  if (tid < BB){
    float acc = bc2[0];
    for (int j = 0; j < 32; ++j) acc += hid[tid * 32 + j] * Wc2[j];
    out[tid] = acc;
  }
}

static inline int cdiv(long a, long b){ return (int)((a + b - 1) / b); }

extern "C" void kernel_launch(void* const* d_in, const int* in_sizes, int n_in,
                              void* d_out, int out_size, void* d_ws, size_t ws_size,
                              hipStream_t stream) {
  (void)in_sizes; (void)n_in; (void)out_size; (void)ws_size;
  const float* x     = (const float*)d_in[0];
  const int*   ei    = (const int*)d_in[1];
  const int*   batch = (const int*)d_in[2];
  const float* W1    = (const float*)d_in[3];
  const float* as1   = (const float*)d_in[4];
  const float* ad1   = (const float*)d_in[5];
  const float* b1    = (const float*)d_in[6];
  const float* W2    = (const float*)d_in[7];
  const float* as2   = (const float*)d_in[8];
  const float* ad2   = (const float*)d_in[9];
  const float* b2    = (const float*)d_in[10];
  const float* Wih   = (const float*)d_in[11];
  const float* Whh   = (const float*)d_in[12];
  const float* bih   = (const float*)d_in[13];
  const float* bhh   = (const float*)d_in[14];
  const float* Wc1   = (const float*)d_in[15];
  const float* bc1   = (const float*)d_in[16];
  const float* Wc2   = (const float*)d_in[17];
  const float* bc2   = (const float*)d_in[18];
  float* out = (float*)d_out;

  // workspace layout (~13.7 MB of f32) — fits L2-resident
  float* ws   = (float*)d_ws;
  const size_t N64 = (size_t)NN * 64;
  float* bufA = ws;                         // node features h  [N,64]
  float* bufB = bufA + N64;                 // aggregation / next-layer input [N,64]
  float* a_s  = bufB + N64;                 // [N,2]
  float* a_d  = a_s + (size_t)NN * 2;       // [N,2]
  float* emax = a_d + (size_t)NN * 2;       // [N,2]
  float* den  = emax + (size_t)NN * 2;      // [N,2]
  float* ebuf = den + (size_t)NN * 2;       // [E+N,2]
  float* emb  = ebuf + (size_t)ETOT * 2;    // [T,B,64]
  float* cnt  = emb + (size_t)TT * BB * 64; // [T,B]
  float* hn   = cnt + (size_t)TT * BB;      // [B,64]

  fill_kernel<<<cdiv(TT * BB * 64, 256), 256, 0, stream>>>(emb, TT * BB * 64, 0.f);
  fill_kernel<<<cdiv(TT * BB, 256), 256, 0, stream>>>(cnt, TT * BB, 0.f);

  const int gemm_blocks = cdiv(NN / 16, 4);

  for (int f = 0; f < TT; ++f){
    const float* xf   = x + (size_t)f * NN * 7;
    const int*   srcf = ei + (size_t)f * 2 * EE;
    const int*   dstf = srcf + EE;
    const int*   batf = batch + (size_t)f * NN;

    // ---- GAT layer 1 (heads=2, C=32) ----
    gemm_wmma_kernel<7><<<gemm_blocks, 128, 0, stream>>>(xf, W1, bufA, NN);
    att_kernel<<<cdiv((long)NN * 2, 256), 256, 0, stream>>>(bufA, as1, ad1, a_s, a_d, NN, 2, 32);
    fill_kernel<<<cdiv((long)NN * 2, 256), 256, 0, stream>>>(emax, NN * 2, -1e30f);
    fill_kernel<<<cdiv((long)NN * 2, 256), 256, 0, stream>>>(den, NN * 2, 0.f);
    fill_kernel<<<cdiv((long)N64, 256), 256, 0, stream>>>(bufB, (int)N64, 0.f);
    edge_max_kernel<<<cdiv((long)ETOT * 2, 256), 256, 0, stream>>>(srcf, dstf, EE, ETOT, a_s, a_d, emax, 2);
    edge_exp_kernel<<<cdiv((long)ETOT * 2, 256), 256, 0, stream>>>(srcf, dstf, EE, ETOT, a_s, a_d, emax, den, ebuf, 2);
    edge_msg_kernel<<<cdiv((long)ETOT * 16, 256), 256, 0, stream>>>(srcf, dstf, EE, ETOT, bufA, ebuf, den, bufB, 2);
    bias_act_kernel<<<cdiv((long)NN * 16, 256), 256, 0, stream>>>(bufB, b1, NN * 16);

    // ---- GAT layer 2 (heads=1, C=64) ----
    gemm_wmma_kernel<64><<<gemm_blocks, 128, 0, stream>>>(bufB, W2, bufA, NN);
    att_kernel<<<cdiv((long)NN, 256), 256, 0, stream>>>(bufA, as2, ad2, a_s, a_d, NN, 1, 64);
    fill_kernel<<<cdiv((long)NN, 256), 256, 0, stream>>>(emax, NN, -1e30f);
    fill_kernel<<<cdiv((long)NN, 256), 256, 0, stream>>>(den, NN, 0.f);
    fill_kernel<<<cdiv((long)N64, 256), 256, 0, stream>>>(bufB, (int)N64, 0.f);
    edge_max_kernel<<<cdiv((long)ETOT, 256), 256, 0, stream>>>(srcf, dstf, EE, ETOT, a_s, a_d, emax, 1);
    edge_exp_kernel<<<cdiv((long)ETOT, 256), 256, 0, stream>>>(srcf, dstf, EE, ETOT, a_s, a_d, emax, den, ebuf, 1);
    edge_msg_kernel<<<cdiv((long)ETOT * 16, 256), 256, 0, stream>>>(srcf, dstf, EE, ETOT, bufA, ebuf, den, bufB, 1);
    bias_act_kernel<<<cdiv((long)NN * 16, 256), 256, 0, stream>>>(bufB, b2, NN * 16);

    // ---- mean pool per graph ----
    pool_cnt_kernel<<<cdiv((long)NN, 256), 256, 0, stream>>>(batf, cnt + (size_t)f * BB, NN);
    pool_sum_kernel<<<cdiv((long)NN * 16, 256), 256, 0, stream>>>(batf, bufB, emb + (size_t)f * BB * 64, NN);
  }

  gru_kernel<<<1, 256, 0, stream>>>(emb, cnt, Wih, Whh, bih, bhh, hn);
  head_kernel<<<1, 512, 0, stream>>>(hn, Wc1, bc1, Wc2, bc2, out);
}
